// JointNetwork_33990371181316
// MI455X (gfx1250) — matrix operand order repaired and checked by reading, not compiled, not run on hardware
//
#include <hip/hip_runtime.h>

// ---------------- CDNA5 (gfx1250) wave32 WMMA bf16 types ----------------
typedef __bf16 bf16_t;
typedef __attribute__((ext_vector_type(16))) __bf16 v16bf;
typedef __attribute__((ext_vector_type(8)))  float  v8f;

union FragBF {
  v16bf v;
  uint4 q[2];
};

__device__ __forceinline__ v8f wmma_bf16(v16bf a, v16bf b, v8f c) {
  // D = A(16x32 bf16) * B(32x16 bf16) + C(16x16 f32)
  return __builtin_amdgcn_wmma_f32_16x16x32_bf16(
      /*neg_a=*/false, a, /*neg_b=*/false, b,
      /*c_mod=*/(short)0, c, /*reuse_a=*/false, /*reuse_b=*/false);
}

// ---------------- Problem constants ----------------
#define B_    4
#define T_    256
#define U_    128
#define DENC  512
#define DPRED 640
#define JD    1024
#define VOUT  1025
#define VPAD  1152      // 9 * 128, zero-padded N for the final GEMM
#define AJ_LD 1032      // 1024 + 8 pad: row stride 516 dwords == 4 (mod 64) -> conflict-free frags
#define AJ_BYTES ((size_t)U_ * AJ_LD * 2)   // 264192 B of the 320 KB WGP LDS

// ---------------- Prep kernels (one-time, tiny traffic) ----------------
__global__ void k_f32_to_bf16(const float* __restrict__ src,
                              bf16_t* __restrict__ dst, int n) {
  int i = blockIdx.x * blockDim.x + threadIdx.x;
  if (i < n) dst[i] = (bf16_t)src[i];
}

// dst[n][k] = (n < cols_src) ? src[k][n] : 0 ; dst is [rows_dst][rows_src] bf16
__global__ void k_transpose_bf16(const float* __restrict__ src,
                                 bf16_t* __restrict__ dst,
                                 int rows_src, int cols_src, int rows_dst) {
  int i = blockIdx.x * blockDim.x + threadIdx.x;
  int total = rows_dst * rows_src;
  if (i >= total) return;
  int n = i / rows_src;
  int k = i - n * rows_src;
  float v = (n < cols_src) ? src[(size_t)k * cols_src + n] : 0.0f;
  dst[i] = (bf16_t)v;
}

// ---------------- Projection GEMM: C[M][N] f32 = A[M][K]bf16 * Bt[N][K]bf16 + bias ----------------
// 256 threads = 8 waves, WG tile 128x128, wave tile 64x32 (4x2 WMMA frags).
__global__ __launch_bounds__(256) void k_proj_gemm(
    const bf16_t* __restrict__ A,
    const bf16_t* __restrict__ Bt,
    const float*  __restrict__ bias,
    float* __restrict__ C,
    int M, int N, int K) {
  const int lane = threadIdx.x & 31;
  const int wid  = threadIdx.x >> 5;
  const int wm   = wid >> 2;      // 0..1
  const int wn   = wid & 3;       // 0..3
  const int l16  = lane & 15;
  const int hh   = lane >> 4;

  const int mbase = blockIdx.y * 128 + wm * 64;
  const int nbase = blockIdx.x * 128 + wn * 32;

  v8f zero = {0.f, 0.f, 0.f, 0.f, 0.f, 0.f, 0.f, 0.f};
  v8f acc[4][2];
#pragma unroll
  for (int f = 0; f < 4; ++f)
#pragma unroll
    for (int g = 0; g < 2; ++g) acc[f][g] = zero;

  for (int kk = 0; kk < K; kk += 32) {
    FragBF bfr[2];
#pragma unroll
    for (int g = 0; g < 2; ++g) {
      const bf16_t* bp = Bt + (size_t)(nbase + g * 16 + l16) * K + kk + hh * 16;
      bfr[g].q[0] = *reinterpret_cast<const uint4*>(bp);       // K +0..7  -> VGPR0..3
      bfr[g].q[1] = *reinterpret_cast<const uint4*>(bp + 8);   // K +8..15 -> VGPR4..7
    }
#pragma unroll
    for (int f = 0; f < 4; ++f) {
      FragBF afr;
      const bf16_t* ap = A + (size_t)(mbase + f * 16 + l16) * K + kk + hh * 16;
      afr.q[0] = *reinterpret_cast<const uint4*>(ap);          // K base..base+7
      afr.q[1] = *reinterpret_cast<const uint4*>(ap + 8);      // K base+8..base+15
#pragma unroll
      for (int g = 0; g < 2; ++g)
        acc[f][g] = wmma_bf16(afr.v, bfr[g].v, acc[f][g]);
    }
  }

#pragma unroll
  for (int g = 0; g < 2; ++g) {
    int col = nbase + g * 16 + l16;
    float bv = bias[col];
#pragma unroll
    for (int f = 0; f < 4; ++f) {
#pragma unroll
      for (int r = 0; r < 8; ++r) {
        int row = mbase + f * 16 + hh * 8 + r;   // C layout: VGPR r -> M = r (+8 for hi half)
        C[(size_t)row * N + col] = acc[f][g][r] + bv;
      }
    }
  }
}

// ---------------- Fused joint GEMM ----------------
// One WG per (b,t). Phase 1: build the FULL 128x1024 bf16 panel
// relu(encP[b,t,:] + predP[b,u,:]) in LDS (predP read exactly once per WG).
// One barrier. Phase 2: barrier-free WMMA sweep; each wave owns TWO disjoint
// 16-column strips (128 apart), so every A-fragment ds_load feeds 2 WMMAs
// (1 ds_load_b128 per WMMA) and each W_outT column is loaded once per WG.
__global__ __launch_bounds__(256) void k_joint_gemm(
    const float*  __restrict__ encP,   // [B*T][1024] f32
    const float*  __restrict__ predP,  // [B*U][1024] f32
    const bf16_t* __restrict__ WoutT,  // [1152][1024] bf16 (zero padded)
    const float*  __restrict__ bias,   // [1025]
    float* __restrict__ out) {         // [B*T*U][1025] f32
  extern __shared__ __attribute__((aligned(16))) bf16_t Aj[];  // [128][AJ_LD]

  const int tid  = threadIdx.x;
  const int lane = tid & 31;
  const int wid  = tid >> 5;      // 0..7
  const int l16  = lane & 15;
  const int hh   = lane >> 4;

  const int bt = blockIdx.x;      // b*T + t
  const int b  = bt >> 8;         // T = 256

  // ---- Phase 1: build joint panel (coalesced 512B/instruction reads) ----
  const float* er       = encP + (size_t)bt * JD;
  const float* predBase = predP + (size_t)b * U_ * JD;
#pragma unroll 1
  for (int r = 0; r < 16; ++r) {
    const int u = wid * 16 + r;
    const float* pr = predBase + (size_t)u * JD;
    bf16_t* arow = Aj + (size_t)u * AJ_LD;
    for (int c = lane * 4; c < JD; c += 128) {
      float4 p = *reinterpret_cast<const float4*>(pr + c);
      float4 e = *reinterpret_cast<const float4*>(er + c);   // L0-resident broadcast row
      __attribute__((aligned(8))) bf16_t t4[4];
      t4[0] = (bf16_t)fmaxf(e.x + p.x, 0.0f);
      t4[1] = (bf16_t)fmaxf(e.y + p.y, 0.0f);
      t4[2] = (bf16_t)fmaxf(e.z + p.z, 0.0f);
      t4[3] = (bf16_t)fmaxf(e.w + p.w, 0.0f);
      *reinterpret_cast<uint2*>(arow + c) = *reinterpret_cast<const uint2*>(t4);  // ds_store_b64
    }
  }
  __syncthreads();   // the ONLY barrier

  const v8f zero = {0.f, 0.f, 0.f, 0.f, 0.f, 0.f, 0.f, 0.f};
  float* const outBase = out + (size_t)bt * U_ * VOUT;

  // ---- Phase 2a: 4 double-strip passes cover columns 0..1023 ----
#pragma unroll 1
  for (int j = 0; j < 4; ++j) {
    const int c0 = j * 256 + wid * 16 + l16;   // lane's first column  (< 1024)
    const int c1 = c0 + 128;                   // lane's second column (< 1024)
    const bf16_t* b0 = WoutT + (size_t)c0 * JD + hh * 16;
    const bf16_t* b1 = b0 + (size_t)128 * JD;

    v8f acc0[8], acc1[8];
#pragma unroll
    for (int f = 0; f < 8; ++f) { acc0[f] = zero; acc1[f] = zero; }

    for (int kk = 0; kk < JD; kk += 32) {
      FragBF f0, f1;
      f0.q[0] = *reinterpret_cast<const uint4*>(b0 + kk);
      f0.q[1] = *reinterpret_cast<const uint4*>(b0 + kk + 8);
      f1.q[0] = *reinterpret_cast<const uint4*>(b1 + kk);
      f1.q[1] = *reinterpret_cast<const uint4*>(b1 + kk + 8);
#pragma unroll
      for (int f = 0; f < 8; ++f) {
        FragBF afr;
        const bf16_t* ap = Aj + (size_t)(f * 16 + l16) * AJ_LD + kk + hh * 16;
        afr.q[0] = *reinterpret_cast<const uint4*>(ap);        // ds_load_b128, bank-clean
        afr.q[1] = *reinterpret_cast<const uint4*>(ap + 8);
        acc0[f] = wmma_bf16(afr.v, f0.v, acc0[f]);             // A-frag reused twice
        acc1[f] = wmma_bf16(afr.v, f1.v, acc1[f]);
      }
    }

    const float bv0 = bias[c0];
    const float bv1 = bias[c1];
    float* o0 = outBase + c0;
#pragma unroll
    for (int f = 0; f < 8; ++f) {
#pragma unroll
      for (int r = 0; r < 8; ++r) {
        const int u = f * 16 + hh * 8 + r;     // C layout: VGPR r -> M = r (+8 hi half)
        o0[(size_t)u * VOUT]       = acc0[f][r] + bv0;
        o0[(size_t)u * VOUT + 128] = acc1[f][r] + bv1;
      }
    }
  }

  // ---- Phase 2b: tail strip, columns 1024..1039 (only 1024 real); wave 0 only ----
  if (wid == 0) {
    const int c = 1024 + l16;
    const bf16_t* bc = WoutT + (size_t)c * JD + hh * 16;  // rows < 1152, zero-padded
    v8f acc[8];
#pragma unroll
    for (int f = 0; f < 8; ++f) acc[f] = zero;

    for (int kk = 0; kk < JD; kk += 32) {
      FragBF bfr;
      bfr.q[0] = *reinterpret_cast<const uint4*>(bc + kk);
      bfr.q[1] = *reinterpret_cast<const uint4*>(bc + kk + 8);
#pragma unroll
      for (int f = 0; f < 8; ++f) {
        FragBF afr;
        const bf16_t* ap = Aj + (size_t)(f * 16 + l16) * AJ_LD + kk + hh * 16;
        afr.q[0] = *reinterpret_cast<const uint4*>(ap);
        afr.q[1] = *reinterpret_cast<const uint4*>(ap + 8);
        acc[f] = wmma_bf16(afr.v, bfr.v, acc[f]);
      }
    }

    if (c < VOUT) {                 // lane guard: only column 1024 stores
      const float bv = bias[c];
#pragma unroll
      for (int f = 0; f < 8; ++f) {
#pragma unroll
        for (int r = 0; r < 8; ++r) {
          const int u = f * 16 + hh * 8 + r;
          outBase[(size_t)u * VOUT + c] = acc[f][r] + bv;
        }
      }
    }
  }
}

// ---------------- Host launcher ----------------
extern "C" void kernel_launch(void* const* d_in, const int* in_sizes, int n_in,
                              void* d_out, int out_size, void* d_ws, size_t ws_size,
                              hipStream_t stream) {
  (void)in_sizes; (void)n_in; (void)out_size; (void)ws_size;
  const float* enc    = (const float*)d_in[0];
  const float* pred   = (const float*)d_in[1];
  const float* W_enc  = (const float*)d_in[2];
  const float* b_enc  = (const float*)d_in[3];
  const float* W_pred = (const float*)d_in[4];
  const float* b_pred = (const float*)d_in[5];
  const float* W_out  = (const float*)d_in[6];
  const float* b_out  = (const float*)d_in[7];
  float* out = (float*)d_out;

  char* ws = (char*)d_ws;
  size_t off = 0;
  auto alloc = [&](size_t bytes) -> void* {
    void* p = ws + off;
    off = (off + bytes + 255) & ~(size_t)255;
    return p;
  };
  bf16_t* encB   = (bf16_t*)alloc((size_t)B_ * T_ * DENC * 2);   // 1.0 MB
  bf16_t* predB  = (bf16_t*)alloc((size_t)B_ * U_ * DPRED * 2);  // 0.64 MB
  bf16_t* WencT  = (bf16_t*)alloc((size_t)JD * DENC * 2);        // 1.0 MB
  bf16_t* WpredT = (bf16_t*)alloc((size_t)JD * DPRED * 2);       // 1.25 MB
  bf16_t* WoutT  = (bf16_t*)alloc((size_t)VPAD * JD * 2);        // 2.25 MB
  float*  encP   = (float*)alloc((size_t)B_ * T_ * JD * 4);      // 4 MB
  float*  predP  = (float*)alloc((size_t)B_ * U_ * JD * 4);      // 2 MB

  // one-time conversions / weight transposes
  int nEnc = B_ * T_ * DENC, nPred = B_ * U_ * DPRED;
  k_f32_to_bf16<<<(nEnc + 255) / 256, 256, 0, stream>>>(enc, encB, nEnc);
  k_f32_to_bf16<<<(nPred + 255) / 256, 256, 0, stream>>>(pred, predB, nPred);
  k_transpose_bf16<<<(JD * DENC + 255) / 256, 256, 0, stream>>>(W_enc, WencT, DENC, JD, JD);
  k_transpose_bf16<<<(JD * DPRED + 255) / 256, 256, 0, stream>>>(W_pred, WpredT, DPRED, JD, JD);
  k_transpose_bf16<<<(VPAD * JD + 255) / 256, 256, 0, stream>>>(W_out, WoutT, JD, VOUT, VPAD);

  // projections (WMMA bf16, f32 accum + bias)
  k_proj_gemm<<<dim3(JD / 128, (B_ * T_) / 128), 256, 0, stream>>>(
      encB, WencT, b_enc, encP, B_ * T_, JD, DENC);
  k_proj_gemm<<<dim3(JD / 128, (B_ * U_) / 128), 256, 0, stream>>>(
      predB, WpredT, b_pred, predP, B_ * U_, JD, DPRED);

  // fused relu(enc+pred) x W_out -> 275 GFLOP, the dominant kernel.
  // One WG per (b,t); 264 KB dynamic LDS holds the full joint panel (320 KB WGP LDS).
  k_joint_gemm<<<dim3(B_ * T_), 256, AJ_BYTES, stream>>>(
      encP, predP, WoutT, b_out, out);
}
